// SSD2d_66640712564975
// MI455X (gfx1250) — compile-verified, matrix operand order checked by reading, not run on hardware
//
#include <hip/hip_runtime.h>
#include <hip/hip_bf16.h>

// ---------------------------------------------------------------------------
// SSD template match: y = 1 - (boxsum(x^2) + sum(w^2) - 2*conv(x,w)) / 196
// x: [8,1,384,384] f32, w: [128,1,7,7] f32, y: [8,128,384,384] f32
// Implicit GEMM via v_wmma_f32_16x16x32_f16 (M=COUT tile 16, N=16 pixels,
// K=49 padded to 64). FP32 accumulate; memory-bound on the 604MB output.
// B (im2col) fragment is block-uniform -> built once in LDS in WMMA lane
// layout, each wave fetches it with aligned ds_load_b128.
// ---------------------------------------------------------------------------

#define BATCH 8
#define HH 384
#define WW 384
#define COUT 128
#define KDIM 49     // 7*7
#define KPAD 64

typedef __attribute__((ext_vector_type(16))) _Float16 v16h;
typedef __attribute__((ext_vector_type(4)))  _Float16 v4h;
typedef __attribute__((ext_vector_type(8)))  float    v8f;
typedef __attribute__((ext_vector_type(4)))  float    v4f;

union AU { v16h v; v4f q[2]; };

// K' stripe owned by lane-group g at fragment slot i (16-bit operand layout,
// ISA 7.12.2): i<8 -> 8g+i ; i>=8 -> 16+8g+(i-8). Same map for A and B.
__device__ __forceinline__ int kslot(int g, int i) {
    return (i < 8) ? (8 * g + i) : (16 + 8 * g + (i - 8));
}

// Pack weights to f16 [COUT][KPAD] (zero-padded K) + per-filter energy sum(w^2).
__global__ void ssd_pack_weights(const float* __restrict__ w,
                                 _Float16* __restrict__ apack,
                                 float* __restrict__ kk) {
    const int t = threadIdx.x;            // one thread per output channel
    if (t >= COUT) return;
    float s = 0.0f;
    #pragma unroll
    for (int k = 0; k < KPAD; ++k) {
        float f = (k < KDIM) ? w[t * KDIM + k] : 0.0f;
        s += f * f;                        // padded entries contribute 0
        apack[t * KPAD + k] = (_Float16)f;
    }
    kk[t] = s;
}

__global__ __launch_bounds__(256) void ssd_main(const float* __restrict__ x,
                                                const _Float16* __restrict__ apack,
                                                const float* __restrict__ kk,
                                                float* __restrict__ out) {
    __shared__ float xt[7][24];            // halo tile rows h-3..h+3, cols w0-3..w0+18
    __shared__ float xs_s[16];             // 7x7 box sum of x^2 per pixel
    __shared__ __align__(32) _Float16 bpack[2][32][16];  // B frags, WMMA lane layout

    const int wt  = blockIdx.x;            // 16-pixel strip along W
    const int h   = blockIdx.y;
    const int b   = blockIdx.z;
    const int w0  = wt * 16;
    const int tid = threadIdx.x;

    // ---- stage 7x22 fp32 halo tile into LDS (zero-padded at borders) ----
    if (tid < 7 * 22) {
        const int r  = tid / 22;
        const int c  = tid % 22;
        const int gh = h - 3 + r;
        const int gw = w0 - 3 + c;
        float v = 0.0f;
        if (gh >= 0 && gh < HH && gw >= 0 && gw < WW)
            v = x[(b * HH + gh) * WW + gw];
        xt[r][c] = v;
    }
    __syncthreads();

    // ---- build the block-uniform B fragment once: 256 threads x 4 halfs ----
    {
        const int s    = tid >> 7;         // K chunk (0/1)
        const int blane = (tid >> 2) & 31; // target lane
        const int i0   = (tid & 3) * 4;    // 4 consecutive fragment slots
        const int g    = blane >> 4;
        const int n    = blane & 15;       // pixel
        v4h vals;
        #pragma unroll
        for (int q = 0; q < 4; ++q) {
            const int i = i0 + q;
            const int k = 32 * s + kslot(g, i);
            const float v = (k < KDIM) ? xt[k / 7][n + (k % 7)] : 0.0f;
            vals[q] = (_Float16)v;
        }
        *(v4h*)&bpack[s][blane][i0] = vals;   // 8B aligned LDS store
    }

    // ---- xs: box sum of squares for the 16 pixels (full fp32) ----
    if (tid < 16) {
        float s = 0.0f;
        #pragma unroll
        for (int dy = 0; dy < 7; ++dy)
            #pragma unroll
            for (int dx = 0; dx < 7; ++dx) {
                const float v = xt[dy][tid + dx];
                s += v * v;
            }
        xs_s[tid] = s;
    }
    __syncthreads();

    // ---- per-wave 16(COUT) x 16(pixel) WMMA tile ----
    const int wave  = tid >> 5;
    const int lane  = tid & 31;
    const int n     = lane & 15;           // pixel / A-matrix row within tile
    const int g     = lane >> 4;           // lane-group selects K sub-stripe
    const int cout0 = wave * 16;

    v8f acc = {};
    #pragma unroll
    for (int s = 0; s < 2; ++s) {          // two K=32 chunks (K padded to 64)
        // A fragment: weight rows, packed so each lane is two 16B loads.
        AU a;
        const _Float16* ab = apack + ((cout0 + n) * KPAD + 32 * s + 8 * g);
        a.q[0] = *(const v4f*)(ab);        // K' = 8g .. 8g+7
        a.q[1] = *(const v4f*)(ab + 16);   // K' = 16+8g .. 16+8g+7

        // B fragment: one aligned 32B LDS read (2x ds_load_b128).
        const v16h bf = *(const v16h*)&bpack[s][lane][0];

        acc = __builtin_amdgcn_wmma_f32_16x16x32_f16(
            /*neg_a=*/false, a.v, /*neg_b=*/false, bf,
            /*c_mod=*/(short)0, acc, /*reuse_a=*/false, /*reuse_b=*/false);
    }

    // ---- fused epilogue: y = 1 - (xs + k - 2c) / 196 ----
    const float xs   = xs_s[n];
    const int   wpix = w0 + n;
    #pragma unroll
    for (int j = 0; j < 8; ++j) {
        const int   cout = cout0 + j + 8 * g;   // C/D layout: VGPR j, lane group g
        const float y = 1.0f - (xs + kk[cout] - 2.0f * acc[j]) * (1.0f / 196.0f);
        out[((b * COUT + cout) * HH + h) * WW + wpix] = y;
    }
}

extern "C" void kernel_launch(void* const* d_in, const int* in_sizes, int n_in,
                              void* d_out, int out_size, void* d_ws, size_t ws_size,
                              hipStream_t stream) {
    const float* x   = (const float*)d_in[0];
    const float* wgt = (const float*)d_in[1];
    float*       out = (float*)d_out;

    _Float16* apack = (_Float16*)d_ws;                                  // 16 KB
    float*    kk    = (float*)((char*)d_ws + COUT * KPAD * sizeof(_Float16));

    ssd_pack_weights<<<1, COUT, 0, stream>>>(wgt, apack, kk);

    dim3 grid(WW / 16, HH, BATCH);   // 24 x 384 x 8 blocks
    ssd_main<<<grid, 256, 0, stream>>>(x, apack, kk, out);
}
